// HybridGNet3D_59700045415096
// MI455X (gfx1250) — matrix operand order repaired and checked by reading, not compiled, not run on hardware
//
#include <hip/hip_runtime.h>

// ---------------------------------------------------------------------------
// CDNA5 (gfx1250) HybridGNet3D forward.
// All GEMM-like work (implicit-GEMM 3D convs, latent linears, ChebConv weight
// contraction, IGSC pos projection) runs on v_wmma_f32_16x16x32_bf16
// (wave32, 16x16 tiles, f32 accumulate). Operand loads are branchless and
// K-contiguous (float2 / global_load_b64 fast path), with register blocking
// (conv: up to 8 M-tiles reuse one im2col B fragment; gemm: up to 4 N-tiles
// reuse one A fragment). Graph ops (scatter, grid-sample, unpool) are
// bandwidth-bound VALU kernels with f32 atomics.
// ---------------------------------------------------------------------------

typedef __attribute__((ext_vector_type(16))) __bf16 v16bf;
typedef __attribute__((ext_vector_type(8)))  float  v8f;

// ----------------------- fragment load helpers ----------------------------
// A fragment (16x32 bf16, contiguous k order): lane holds rows m=lane&15; elements
// 0..15 = K[k0+16*half .. +16).
// p must point at row + k0 + 16*half, 8-byte aligned.
__device__ __forceinline__ v16bf load_a_frag_fast(const float* __restrict__ p)
{
  v16bf f;
#pragma unroll
  for (int i = 0; i < 4; ++i) {
    float2 q = *(const float2*)(p + 2 * i);
    f[2 * i] = (__bf16)q.x; f[2 * i + 1] = (__bf16)q.y;
  }
#pragma unroll
  for (int i = 0; i < 4; ++i) {
    float2 q = *(const float2*)(p + 8 + 2 * i);
    f[8 + 2 * i] = (__bf16)q.x; f[9 + 2 * i] = (__bf16)q.y;
  }
  return f;
}

// scalar (clamped + masked) A fragment for K-tails / odd strides
__device__ __forceinline__ v16bf load_a_frag_tail(const float* __restrict__ prow,
                                                  int k0, int half, int K)
{
  v16bf f;
#pragma unroll
  for (int e = 0; e < 16; ++e) {
    int kk = e + 16 * half;   // one run of 16
    int k = k0 + kk;
    float x = prow[min(k, K - 1)];
    f[e] = (__bf16)(k < K ? x : 0.f);
  }
  return f;
}

// B fragment (32x16 bf16): lane holds col n=lane&15; elements 0..15 =
// K[k0+16*half .. +16). p at col-row + k0 + 16*half, 8-byte aligned.
__device__ __forceinline__ v16bf load_b_frag_fast(const float* __restrict__ p)
{
  v16bf f;
#pragma unroll
  for (int i = 0; i < 8; ++i) {
    float2 q = *(const float2*)(p + 2 * i);
    f[2 * i] = (__bf16)q.x; f[2 * i + 1] = (__bf16)q.y;
  }
  return f;
}

__device__ __forceinline__ v16bf load_b_frag_tail(const float* __restrict__ prow,
                                                  int k0, int half, int K)
{
  v16bf f;
#pragma unroll
  for (int e = 0; e < 16; ++e) {
    int k = k0 + 16 * half + e;
    float x = prow[min(k, K - 1)];
    f[e] = (__bf16)(k < K ? x : 0.f);
  }
  return f;
}

// ======================= generic WMMA GEMM ================================
// C[M,N] = A(M,K; leading dim lda) * BT(N,K; leading dim ldb)^T
// (+bias[n]) (+relu). C compact row-major. lda/ldb must be even (8B align).
// Each wave computes 16(M) x 16*NBLK(N), reusing one A fragment per chunk.
template<int NBLK>
__global__ void gemm_bf16_wmma_k(const float* __restrict__ A, long lda,
                                 const float* __restrict__ BT, long ldb,
                                 float* __restrict__ C, int M, int N, int K,
                                 const float* __restrict__ bias, int relu)
{
  int wave = blockIdx.x * (blockDim.x >> 5) + (threadIdx.x >> 5);
  int lane = threadIdx.x & 31;
  int half = lane >> 4, l15 = lane & 15;
  int ntn = (N + 16 * NBLK - 1) / (16 * NBLK);
  int ntm = (M + 15) >> 4;
  if (wave >= ntm * ntn) return;
  int tm = wave / ntn, tn = wave % ntn;
  int m0 = tm << 4, n0 = tn * 16 * NBLK;

  int am = m0 + l15;
  const float* pa = A + (long)min(am, M - 1) * lda;   // OOB rows: garbage, discarded
  const float* pb[NBLK];
#pragma unroll
  for (int j = 0; j < NBLK; ++j) {
    int bn = n0 + 16 * j + l15;
    pb[j] = BT + (long)min(bn, N - 1) * ldb;
  }

  v8f acc[NBLK];
#pragma unroll
  for (int j = 0; j < NBLK; ++j) acc[j] = (v8f){};

  int kmain = K & ~31;
  for (int k0 = 0; k0 < kmain; k0 += 32) {            // fast path: vector loads
    v16bf af = load_a_frag_fast(pa + k0 + 16 * half);
#pragma unroll
    for (int j = 0; j < NBLK; ++j) {
      v16bf bfr = load_b_frag_fast(pb[j] + k0 + 16 * half);
      acc[j] = __builtin_amdgcn_wmma_f32_16x16x32_bf16(false, af, false, bfr,
                                                       (short)0, acc[j], false, false);
    }
  }
  if (kmain < K) {                                     // masked tail (uniform branch)
    v16bf af = load_a_frag_tail(pa, kmain, half, K);
#pragma unroll
    for (int j = 0; j < NBLK; ++j) {
      v16bf bfr = load_b_frag_tail(pb[j], kmain, half, K);
      acc[j] = __builtin_amdgcn_wmma_f32_16x16x32_bf16(false, af, false, bfr,
                                                       (short)0, acc[j], false, false);
    }
  }

#pragma unroll
  for (int j = 0; j < NBLK; ++j)
#pragma unroll
    for (int r = 0; r < 8; ++r) {
      int m = m0 + r + 8 * half;
      int n = n0 + 16 * j + l15;
      if (m < M && n < N) {
        float v = acc[j][r];
        if (bias) v += bias[n];
        if (relu) v = fmaxf(v, 0.f);
        C[(long)m * N + n] = v;
      }
    }
}

// ================= implicit-GEMM 3D conv (SAME, stride 1) =================
// M = Cout, N = B*D*H*W, K = Cin*KS^3. Wave computes 16*MBLK (M) x 16 (N);
// the expensive im2col B gather is reused for MBLK WMMAs. Halo/OOB values
// are zeroed by select on clamped (always-valid) addresses — no exec dance.
template<int KS>
__device__ __forceinline__ v16bf conv_b_frag(const float* __restrict__ X,
                                             long DHW, long HW,
                                             int b, int Cin, int D, int H, int W,
                                             int z, int yy, int xx,
                                             int k0, int half, int K, bool nok)
{
  const int KS3 = KS * KS * KS;
  const int pad = KS / 2;
  v16bf f;
#pragma unroll
  for (int e = 0; e < 16; ++e) {
    int bk = k0 + 16 * half + e;
    int bkc = min(bk, K - 1);
    int ci = bkc / KS3; int rr = bkc % KS3;
    int kd = rr / (KS * KS); int r3 = rr % (KS * KS);
    int kh = r3 / KS; int kw = r3 % KS;
    int iz = z + kd - pad, iy = yy + kh - pad, ix = xx + kw - pad;
    bool ok = nok && (bk < K) && iz >= 0 && iz < D && iy >= 0 && iy < H &&
              ix >= 0 && ix < W;
    int izc = min(max(iz, 0), D - 1);
    int iyc = min(max(iy, 0), H - 1);
    int ixc = min(max(ix, 0), W - 1);
    float x = X[((long)(b * Cin + ci)) * DHW + (long)izc * HW + (long)iyc * W + ixc];
    f[e] = (__bf16)(ok ? x : 0.f);
  }
  return f;
}

template<int KS, int MBLK>
__global__ void conv3d_bf16_wmma_k(const float* __restrict__ X,
                                   const float* __restrict__ Wt,
                                   const float* __restrict__ Add,
                                   float* __restrict__ Y,
                                   int B, int Cin, int D, int H, int W,
                                   int Cout, int relu)
{
  const int KS3 = KS * KS * KS;
  long wave = (long)blockIdx.x * (blockDim.x >> 5) + (threadIdx.x >> 5);
  int lane = threadIdx.x & 31;
  int half = lane >> 4, l15 = lane & 15;
  long HW  = (long)H * W;
  long DHW = (long)D * HW;
  long Nsp = (long)B * DHW;
  int  K   = Cin * KS3;
  long ntn = (Nsp + 15) >> 4;
  int  ntm = (Cout + 16 * MBLK - 1) / (16 * MBLK);
  if (wave >= (long)ntm * ntn) return;
  int  tm = (int)(wave / ntn);
  long tn = wave % ntn;
  int  m0 = tm * 16 * MBLK;
  long n0 = tn << 4;

  long n  = n0 + l15;
  bool nok = (n < Nsp);
  long nc = nok ? n : (Nsp - 1);
  int b = (int)(nc / DHW); long r1 = nc % DHW;
  int z = (int)(r1 / HW);  long r2 = r1 % HW;
  int yy = (int)(r2 / W);
  int xx = (int)(r2 % W);

  const float* pw[MBLK];
#pragma unroll
  for (int j = 0; j < MBLK; ++j) {
    int amj = m0 + 16 * j + l15;
    pw[j] = Wt + (long)min(amj, Cout - 1) * K;
  }

  v8f acc[MBLK];
#pragma unroll
  for (int j = 0; j < MBLK; ++j) acc[j] = (v8f){};

  bool keven = ((K & 1) == 0);
  for (int k0 = 0; k0 < K; k0 += 32) {
    v16bf bfr = conv_b_frag<KS>(X, DHW, HW, b, Cin, D, H, W, z, yy, xx,
                                k0, half, K, nok);
    if (keven && (k0 + 32 <= K)) {                    // uniform branch
#pragma unroll
      for (int j = 0; j < MBLK; ++j) {
        v16bf af = load_a_frag_fast(pw[j] + k0 + 16 * half);
        acc[j] = __builtin_amdgcn_wmma_f32_16x16x32_bf16(false, af, false, bfr,
                                                         (short)0, acc[j], false, false);
      }
    } else {
#pragma unroll
      for (int j = 0; j < MBLK; ++j) {
        v16bf af = load_a_frag_tail(pw[j], k0, half, K);
        acc[j] = __builtin_amdgcn_wmma_f32_16x16x32_bf16(false, af, false, bfr,
                                                         (short)0, acc[j], false, false);
      }
    }
  }

#pragma unroll
  for (int j = 0; j < MBLK; ++j)
#pragma unroll
    for (int r = 0; r < 8; ++r) {
      int m = m0 + 16 * j + r + 8 * half;
      if (m < Cout && nok) {
        long idx = ((long)(b * Cout + m)) * DHW + (long)z * HW + (long)yy * W + xx;
        float v = acc[j][r];
        if (Add) v += Add[idx];
        if (relu) v = fmaxf(v, 0.f);
        Y[idx] = v;
      }
    }
}

// =========================== elementwise kernels ==========================
__global__ void maxpool3d_k(const float* __restrict__ X, float* __restrict__ Y,
                            int B, int C, int D, int H, int W,
                            int wd, int wh, int ww)
{
  int Do = D / wd, Ho = H / wh, Wo = W / ww;
  long total = (long)B * C * Do * Ho * Wo;
  long t = (long)blockIdx.x * blockDim.x + threadIdx.x;
  if (t >= total) return;
  int xo = (int)(t % Wo); long r = t / Wo;
  int yo = (int)(r % Ho); r /= Ho;
  int zo = (int)(r % Do); r /= Do;
  int c  = (int)(r % C);  int b = (int)(r / C);
  const float* base = X + ((long)(b * C + c)) * D * H * W;
  float m = -3.4e38f;
  for (int dz = 0; dz < wd; ++dz)
    for (int dy = 0; dy < wh; ++dy)
      for (int dx = 0; dx < ww; ++dx)
        m = fmaxf(m, base[(long)(zo * wd + dz) * H * W +
                          (long)(yo * wh + dy) * W + (xo * ww + dx)]);
  Y[t] = m;
}

__global__ void zero_k(float* p, long n)
{
  long t = (long)blockIdx.x * blockDim.x + threadIdx.x;
  if (t < n) p[t] = 0.f;
}

// dst[c][r] = src[r][c]   (src R x Cc row-major)
__global__ void transpose_k(const float* __restrict__ src, float* __restrict__ dst,
                            int R, int Cc)
{
  long t = (long)blockIdx.x * blockDim.x + threadIdx.x;
  if (t >= (long)R * Cc) return;
  int r = (int)(t / Cc), c = (int)(t % Cc);
  dst[(long)c * R + r] = src[(long)r * Cc + c];
}

__global__ void deg_k(const int* __restrict__ row, float* __restrict__ deg, int E)
{
  int e = blockIdx.x * blockDim.x + threadIdx.x;
  if (e < E) atomicAdd(&deg[row[e]], 1.f);
}

__global__ void norm_k(const int* __restrict__ row, const int* __restrict__ col,
                       const float* __restrict__ deg, float* __restrict__ nrm, int E)
{
  int e = blockIdx.x * blockDim.x + threadIdx.x;
  if (e >= E) return;
  float dr = deg[row[e]], dc = deg[col[e]];
  float ir = (dr > 0.f) ? rsqrtf(fmaxf(dr, 1.f)) : 0.f;
  float ic = (dc > 0.f) ? rsqrtf(fmaxf(dc, 1.f)) : 0.f;
  nrm[e] = -(ir * ic);
}

// dst[r*dstStride + dstOff + c] = src[r*srcStride + c]
__global__ void copy_cols_k(float* __restrict__ dst, int dstStride, int dstOff,
                            const float* __restrict__ src, int srcStride,
                            long rows, int cols)
{
  long t = (long)blockIdx.x * blockDim.x + threadIdx.x;
  if (t >= rows * cols) return;
  long r = t / cols; int c = (int)(t % cols);
  dst[r * dstStride + dstOff + c] = src[r * srcStride + c];
}

// T[row, kdst*C + c] = alpha * T[row, ksrc*C + c]
__global__ void cheb_init_k(float* __restrict__ T, long BN, int C, int KC,
                            int kdst, int ksrc, float alpha)
{
  long t = (long)blockIdx.x * blockDim.x + threadIdx.x;
  if (t >= BN * C) return;
  long r = t / C; int c = (int)(t % C);
  T[r * KC + (long)kdst * C + c] = alpha * T[r * KC + (long)ksrc * C + c];
}

// T[b, col[e], kdst, c] += scale*norm[e] * T[b, row[e], ksrc, c]
__global__ void cheb_scatter_k(const int* __restrict__ row, const int* __restrict__ col,
                               const float* __restrict__ nrm, float* __restrict__ T,
                               int Bn, int Nn, int E, int C, int KC,
                               int kdst, int ksrc, float scale)
{
  long t = (long)blockIdx.x * blockDim.x + threadIdx.x;
  long tot = (long)Bn * E * C;
  if (t >= tot) return;
  int c = (int)(t % C); long r = t / C;
  int e = (int)(r % E); int b = (int)(r / E);
  float v = T[((long)b * Nn + row[e]) * KC + (long)ksrc * C + c] * (scale * nrm[e]);
  atomicAdd(&T[((long)b * Nn + col[e]) * KC + (long)kdst * C + c], v);
}

__global__ void instnorm_relu_k(float* __restrict__ X, long BN, int C)
{
  long t = (long)blockIdx.x * blockDim.x + threadIdx.x;
  if (t >= BN) return;
  float* r = X + t * C;
  float mu = 0.f;
  for (int c = 0; c < C; ++c) mu += r[c];
  mu /= C;
  float var = 0.f;
  for (int c = 0; c < C; ++c) { float d = r[c] - mu; var += d * d; }
  var /= C;
  float is = rsqrtf(var + 1e-5f);
  for (int c = 0; c < C; ++c) r[c] = fmaxf((r[c] - mu) * is, 0.f);
}

// trilinear grid-sample (align_corners=True, zero padding) into concat cols
__global__ void grid_sample3d_k(const float* __restrict__ feat,
                                const float* __restrict__ pos,
                                float* __restrict__ dst,
                                int B, int C, int D, int H, int W,
                                int N, int dstStride, int dstOff)
{
  long t = (long)blockIdx.x * blockDim.x + threadIdx.x;
  if (t >= (long)B * N) return;
  int b = (int)(t / N), n = (int)(t % N);
  const float* p = pos + ((long)b * N + n) * 3;
  float px = 2.f * p[0] - 1.f, py = 2.f * p[1] - 1.f, pz = 2.f * p[2] - 1.f;
  float gx = (px + 1.f) * 0.5f * (W - 1);
  float gy = (py + 1.f) * 0.5f * (H - 1);
  float gz = (pz + 1.f) * 0.5f * (D - 1);
  float x0 = floorf(gx), y0 = floorf(gy), z0 = floorf(gz);
  float* out = dst + ((long)b * N + n) * dstStride + dstOff;
  long HW = (long)H * W, DHW = (long)D * HW;
  for (int c = 0; c < C; ++c) {
    const float* fb = feat + ((long)b * C + c) * DHW;
    float acc = 0.f;
#pragma unroll
    for (int dz = 0; dz < 2; ++dz)
#pragma unroll
      for (int dy = 0; dy < 2; ++dy)
#pragma unroll
        for (int dx = 0; dx < 2; ++dx) {
          float xi = x0 + dx, yi = y0 + dy, zi = z0 + dz;
          float w = (1.f - fabsf(gx - xi)) * (1.f - fabsf(gy - yi)) * (1.f - fabsf(gz - zi));
          bool valid = (xi >= 0.f) && (xi <= W - 1) && (yi >= 0.f) && (yi <= H - 1) &&
                       (zi >= 0.f) && (zi <= D - 1);
          int xc = min(max((int)xi, 0), W - 1);
          int yc = min(max((int)yi, 0), H - 1);
          int zc = min(max((int)zi, 0), D - 1);
          acc += valid ? fb[(long)zc * HW + (long)yc * W + xc] * w : 0.f;
        }
    out[c] = acc;
  }
}

// y[b,no,c] = sum_{k<3} x[b, idx[no,k], c] * w[no,k]
__global__ void unpool_k(const float* __restrict__ x, const int* __restrict__ idx,
                         const float* __restrict__ w, float* __restrict__ y,
                         int Bn, int No, int Ni, int C)
{
  long t = (long)blockIdx.x * blockDim.x + threadIdx.x;
  long tot = (long)Bn * No * C;
  if (t >= tot) return;
  int c = (int)(t % C); long r = t / C;
  int no = (int)(r % No); int b = (int)(r / No);
  float acc = 0.f;
#pragma unroll
  for (int k = 0; k < 3; ++k)
    acc += x[((long)b * Ni + idx[no * 3 + k]) * C + c] * w[no * 3 + k];
  y[((long)b * No + no) * C + c] = acc;
}

// ============================ host helpers ================================
static inline unsigned eb(long n) { return (unsigned)((n + 255) / 256); }

// A (M,K) lda; BT (N,K) ldb. lda/ldb must be even.
static void launch_gemm(hipStream_t s, const float* A, long lda,
                        const float* BT, long ldb, float* C,
                        int M, int N, int K, const float* bias, int relu)
{
  int nblk = (N >= 64) ? 4 : (N >= 32 ? 2 : 1);
  long ntm = (M + 15) / 16;
  long ntn = (N + 16 * nblk - 1) / (16 * nblk);
  int blocks = (int)((ntm * ntn + 3) / 4);
  if (nblk == 4)
    gemm_bf16_wmma_k<4><<<blocks, 128, 0, s>>>(A, lda, BT, ldb, C, M, N, K, bias, relu);
  else if (nblk == 2)
    gemm_bf16_wmma_k<2><<<blocks, 128, 0, s>>>(A, lda, BT, ldb, C, M, N, K, bias, relu);
  else
    gemm_bf16_wmma_k<1><<<blocks, 128, 0, s>>>(A, lda, BT, ldb, C, M, N, K, bias, relu);
}

template<int KS>
static void launch_conv(hipStream_t s, const float* X, const float* W, const float* Add,
                        float* Y, int B, int Cin, int D, int H, int Wd, int Cout, int relu)
{
  long Nsp = (long)B * D * H * Wd;
  long ntn = (Nsp + 15) / 16;
  int mblk = (Cout >= 128) ? 8 : (Cout >= 64) ? 4 : (Cout >= 32) ? 2 : 1;
  long ntm = (Cout + 16 * mblk - 1) / (16 * mblk);
  int blocks = (int)((ntm * ntn + 3) / 4);
  if (mblk == 8)
    conv3d_bf16_wmma_k<KS, 8><<<blocks, 128, 0, s>>>(X, W, Add, Y, B, Cin, D, H, Wd, Cout, relu);
  else if (mblk == 4)
    conv3d_bf16_wmma_k<KS, 4><<<blocks, 128, 0, s>>>(X, W, Add, Y, B, Cin, D, H, Wd, Cout, relu);
  else if (mblk == 2)
    conv3d_bf16_wmma_k<KS, 2><<<blocks, 128, 0, s>>>(X, W, Add, Y, B, Cin, D, H, Wd, Cout, relu);
  else
    conv3d_bf16_wmma_k<KS, 1><<<blocks, 128, 0, s>>>(X, W, Add, Y, B, Cin, D, H, Wd, Cout, relu);
}

static void launch_pool(hipStream_t s, const float* X, float* Y,
                        int B, int C, int D, int H, int W, int wd, int wh, int ww)
{
  long total = (long)B * C * (D / wd) * (H / wh) * (W / ww);
  maxpool3d_k<<<eb(total), 256, 0, s>>>(X, Y, B, C, D, H, W, wd, wh, ww);
}

// ChebConv K=6: T[b,n,k,c] recurrence, then one WMMA GEMM with W^T (Cout,6C)
static void run_cheb(hipStream_t s, const float* x, int Bn, int Nn, int C,
                     const int* edge, int E, const float* Wg, const float* bias,
                     int Cout, float* T, float* out, float* deg, float* nrm,
                     float* wgt /* Cout x 6C transpose arena */)
{
  const int Kc = 6, KC = Kc * C;
  const int* row = edge;
  const int* col = edge + E;
  long BN = (long)Bn * Nn;
  transpose_k<<<eb((long)KC * Cout), 256, 0, s>>>(Wg, wgt, KC, Cout);
  zero_k<<<eb(Nn), 256, 0, s>>>(deg, Nn);
  deg_k<<<eb(E), 256, 0, s>>>(row, deg, E);
  norm_k<<<eb(E), 256, 0, s>>>(row, col, deg, nrm, E);
  // T0 = x
  copy_cols_k<<<eb(BN * C), 256, 0, s>>>(T, KC, 0, x, C, BN, C);
  // T1 = S(x)
  cheb_init_k<<<eb(BN * C), 256, 0, s>>>(T, BN, C, KC, 1, 0, 0.f);
  cheb_scatter_k<<<eb((long)Bn * E * C), 256, 0, s>>>(row, col, nrm, T, Bn, Nn, E, C, KC, 1, 0, 1.f);
  // Tk = 2*S(T_{k-1}) - T_{k-2}
  for (int k = 2; k < Kc; ++k) {
    cheb_init_k<<<eb(BN * C), 256, 0, s>>>(T, BN, C, KC, k, k - 2, -1.f);
    cheb_scatter_k<<<eb((long)Bn * E * C), 256, 0, s>>>(row, col, nrm, T, Bn, Nn, E, C, KC, k, k - 1, 2.f);
  }
  // out[(b,n), f] = sum_{kc} T[(b,n), kc] * Wg[kc, f]
  launch_gemm(s, T, KC, wgt, KC, out, (int)BN, Cout, KC, bias, 0);
}

// ============================ orchestration ===============================
extern "C" void kernel_launch(void* const* d_in, const int* in_sizes, int n_in,
                              void* d_out, int out_size, void* d_ws, size_t ws_size,
                              hipStream_t stream)
{
  (void)in_sizes; (void)n_in; (void)out_size; (void)ws_size;
  const float* sax = (const float*)d_in[0];
  const float *w1[7], *w2[7], *wsk[7];
  int ii = 1;
  for (int i = 1; i <= 6; ++i) {
    w1[i] = (const float*)d_in[ii++];
    w2[i] = (const float*)d_in[ii++];
    wsk[i] = (i < 6) ? (const float*)d_in[ii++] : nullptr;
  }
  const float* mu_w = (const float*)d_in[18];
  const float* mu_b = (const float*)d_in[19];
  const float* ld_w = (const float*)d_in[22];
  const float* ld_b = (const float*)d_in[23];
  const float *gcw[5], *gcb[5];
  for (int i = 0; i < 5; ++i) { gcw[i] = (const float*)d_in[24 + 2 * i]; gcb[i] = (const float*)d_in[25 + 2 * i]; }
  const float* gcout_w = (const float*)d_in[34];
  const float* igw[4]; for (int i = 0; i < 4; ++i) igw[i] = (const float*)d_in[35 + i];
  const float* upw[4]; for (int i = 0; i < 4; ++i) upw[i] = (const float*)d_in[39 + i];
  const int* edge[5];  for (int i = 0; i < 5; ++i) edge[i] = (const int*)d_in[43 + i];
  const int* upidx[4]; for (int i = 0; i < 4; ++i) upidx[i] = (const int*)d_in[48 + i];

  float* ws = (float*)d_ws;
  float* buf0 = ws;
  float* buf1 = ws + 19000000L;
  float* buf2 = ws + 38000000L;
  float* l3   = ws + 57000000L;   //  4,718,592
  float* l4   = ws + 61800000L;   //  1,179,648
  float* l5   = ws + 63000000L;   //    589,824
  float* l6   = ws + 63600000L;   //    147,456
  float* muv  = ws + 63800000L;   //        256
  float* deg  = ws + 63802000L;   //     11,000
  float* nrm  = ws + 63814000L;   //     66,000
  float* wgt  = ws + 63900000L;   // <= 74,880 (transposed cheb weights)
  float* igt  = ws + 63980000L;   // <=    192 (transposed igsc weights)
  float* out  = (float*)d_out;
  const int Bn = 4;

  // -------------------------- Encoder3D -----------------------------------
  // block1: 1 -> 8 @ (16,192,192)
  launch_conv<3>(stream, sax, w1[1], nullptr, buf0, Bn, 1, 16, 192, 192, 8, 1);
  launch_conv<1>(stream, sax, wsk[1], nullptr, buf1, Bn, 1, 16, 192, 192, 8, 0);
  launch_conv<3>(stream, buf0, w2[1], buf1, buf2, Bn, 8, 16, 192, 192, 8, 1);
  launch_pool(stream, buf2, buf0, Bn, 8, 16, 192, 192, 1, 2, 2);
  // block2: 8 -> 16 @ (16,96,96)
  launch_conv<3>(stream, buf0, w1[2], nullptr, buf1, Bn, 8, 16, 96, 96, 16, 1);
  launch_conv<1>(stream, buf0, wsk[2], nullptr, buf2, Bn, 8, 16, 96, 96, 16, 0);
  launch_conv<3>(stream, buf1, w2[2], buf2, buf0, Bn, 16, 16, 96, 96, 16, 1);
  launch_pool(stream, buf0, buf1, Bn, 16, 16, 96, 96, 1, 2, 2);
  // block3: 16 -> 32 @ (16,48,48) -> keep l3
  launch_conv<3>(stream, buf1, w1[3], nullptr, buf0, Bn, 16, 16, 48, 48, 32, 1);
  launch_conv<1>(stream, buf1, wsk[3], nullptr, buf2, Bn, 16, 16, 48, 48, 32, 0);
  launch_conv<3>(stream, buf0, w2[3], buf2, l3, Bn, 32, 16, 48, 48, 32, 1);
  launch_pool(stream, l3, buf0, Bn, 32, 16, 48, 48, 2, 2, 2);
  // block4: 32 -> 64 @ (8,24,24) -> keep l4
  launch_conv<3>(stream, buf0, w1[4], nullptr, buf1, Bn, 32, 8, 24, 24, 64, 1);
  launch_conv<1>(stream, buf0, wsk[4], nullptr, buf2, Bn, 32, 8, 24, 24, 64, 0);
  launch_conv<3>(stream, buf1, w2[4], buf2, l4, Bn, 64, 8, 24, 24, 64, 1);
  launch_pool(stream, l4, buf0, Bn, 64, 8, 24, 24, 1, 2, 2);
  // block5: 64 -> 128 @ (8,12,12) -> keep l5
  launch_conv<3>(stream, buf0, w1[5], nullptr, buf1, Bn, 64, 8, 12, 12, 128, 1);
  launch_conv<1>(stream, buf0, wsk[5], nullptr, buf2, Bn, 64, 8, 12, 12, 128, 0);
  launch_conv<3>(stream, buf1, w2[5], buf2, l5, Bn, 128, 8, 12, 12, 128, 1);
  launch_pool(stream, l5, buf0, Bn, 128, 8, 12, 12, 1, 2, 2);   // buf0 = pre-res6 "x"
  // block6: 128 -> 128 @ (8,6,6), identity skip -> keep l6
  launch_conv<3>(stream, buf0, w1[6], nullptr, buf1, Bn, 128, 8, 6, 6, 128, 1);
  launch_conv<3>(stream, buf1, w2[6], buf0, l6, Bn, 128, 8, 6, 6, 128, 1);

  // -------------------------- latent ---------------------------------------
  // mu = flat @ mu_w.T + mu_b  (flat = buf0 as (4,36864); mu_w is (64,36864) = BT)
  launch_gemm(stream, buf0, 36864, mu_w, 36864, muv, 4, 64, 36864, mu_b, 0);
  // x = relu(mu @ ld_w.T + ld_b) -> (4,688,128) in buf0 ; ld_w (88064,64) = BT
  launch_gemm(stream, muv, 64, ld_w, 64, buf0, 4, 88064, 64, ld_b, 1);

  // -------------------------- decoder --------------------------------------
  // ---- level 1: N=688, 128 -> 64, edge4, feat=l6(128,8,6,6) ----
  {
    int Nn = 688, C = 128, Co = 64, E = 6 * Nn;
    run_cheb(stream, buf0, Bn, Nn, C, edge[4], E, gcw[0], gcb[0], Co, buf1, buf2, deg, nrm, wgt);
    instnorm_relu_k<<<eb((long)Bn * Nn), 256, 0, stream>>>(buf2, (long)Bn * Nn, Co);
    float* pos = out + 247500;                                   // ds1
    transpose_k<<<eb((long)Co * 3), 256, 0, stream>>>(igw[0], igt, Co, 3);
    launch_gemm(stream, buf2, Co, igt, Co, pos, Bn * Nn, 3, Co, nullptr, 0);
    int Cf = 128, CT = Co + Cf + 3;                              // 195
    copy_cols_k<<<eb((long)Bn * Nn * Co), 256, 0, stream>>>(buf0, CT, 0, buf2, Co, (long)Bn * Nn, Co);
    grid_sample3d_k<<<eb((long)Bn * Nn), 256, 0, stream>>>(l6, pos, buf0, Bn, Cf, 8, 6, 6, Nn, CT, Co);
    copy_cols_k<<<eb((long)Bn * Nn * 3), 256, 0, stream>>>(buf0, CT, Co + Cf, pos, 3, (long)Bn * Nn, 3);
    unpool_k<<<eb((long)Bn * 1375 * CT), 256, 0, stream>>>(buf0, upidx[3], upw[3], buf1, Bn, 1375, Nn, CT);
  }
  // ---- level 2: N=1375, 195 -> 64, edge3, feat=l5(128,8,12,12) ----
  {
    int Nn = 1375, C = 195, Co = 64, E = 6 * Nn;
    run_cheb(stream, buf1, Bn, Nn, C, edge[3], E, gcw[1], gcb[1], Co, buf2, buf0, deg, nrm, wgt);
    instnorm_relu_k<<<eb((long)Bn * Nn), 256, 0, stream>>>(buf0, (long)Bn * Nn, Co);
    float* pos = out + 231000;                                   // ds2
    transpose_k<<<eb((long)Co * 3), 256, 0, stream>>>(igw[1], igt, Co, 3);
    launch_gemm(stream, buf0, Co, igt, Co, pos, Bn * Nn, 3, Co, nullptr, 0);
    int Cf = 128, CT = Co + Cf + 3;                              // 195
    copy_cols_k<<<eb((long)Bn * Nn * Co), 256, 0, stream>>>(buf2, CT, 0, buf0, Co, (long)Bn * Nn, Co);
    grid_sample3d_k<<<eb((long)Bn * Nn), 256, 0, stream>>>(l5, pos, buf2, Bn, Cf, 8, 12, 12, Nn, CT, Co);
    copy_cols_k<<<eb((long)Bn * Nn * 3), 256, 0, stream>>>(buf2, CT, Co + Cf, pos, 3, (long)Bn * Nn, 3);
    unpool_k<<<eb((long)Bn * 2750 * CT), 256, 0, stream>>>(buf2, upidx[2], upw[2], buf0, Bn, 2750, Nn, CT);
  }
  // ---- level 3: N=2750, 195 -> 32, edge2, feat=l4(64,8,24,24) ----
  {
    int Nn = 2750, C = 195, Co = 32, E = 6 * Nn;
    run_cheb(stream, buf0, Bn, Nn, C, edge[2], E, gcw[2], gcb[2], Co, buf1, buf2, deg, nrm, wgt);
    instnorm_relu_k<<<eb((long)Bn * Nn), 256, 0, stream>>>(buf2, (long)Bn * Nn, Co);
    float* pos = out + 198000;                                   // ds3
    transpose_k<<<eb((long)Co * 3), 256, 0, stream>>>(igw[2], igt, Co, 3);
    launch_gemm(stream, buf2, Co, igt, Co, pos, Bn * Nn, 3, Co, nullptr, 0);
    int Cf = 64, CT = Co + Cf + 3;                               // 99
    copy_cols_k<<<eb((long)Bn * Nn * Co), 256, 0, stream>>>(buf1, CT, 0, buf2, Co, (long)Bn * Nn, Co);
    grid_sample3d_k<<<eb((long)Bn * Nn), 256, 0, stream>>>(l4, pos, buf1, Bn, Cf, 8, 24, 24, Nn, CT, Co);
    copy_cols_k<<<eb((long)Bn * Nn * 3), 256, 0, stream>>>(buf1, CT, Co + Cf, pos, 3, (long)Bn * Nn, 3);
    unpool_k<<<eb((long)Bn * 5500 * CT), 256, 0, stream>>>(buf1, upidx[1], upw[1], buf0, Bn, 5500, Nn, CT);
  }
  // ---- level 4: N=5500, 99 -> 32, edge1, feat=l3(32,16,48,48) ----
  {
    int Nn = 5500, C = 99, Co = 32, E = 6 * Nn;
    run_cheb(stream, buf0, Bn, Nn, C, edge[1], E, gcw[3], gcb[3], Co, buf1, buf2, deg, nrm, wgt);
    instnorm_relu_k<<<eb((long)Bn * Nn), 256, 0, stream>>>(buf2, (long)Bn * Nn, Co);
    float* pos = out + 132000;                                   // ds4
    transpose_k<<<eb((long)Co * 3), 256, 0, stream>>>(igw[3], igt, Co, 3);
    launch_gemm(stream, buf2, Co, igt, Co, pos, Bn * Nn, 3, Co, nullptr, 0);
    int Cf = 32, CT = Co + Cf + 3;                               // 67
    copy_cols_k<<<eb((long)Bn * Nn * Co), 256, 0, stream>>>(buf1, CT, 0, buf2, Co, (long)Bn * Nn, Co);
    grid_sample3d_k<<<eb((long)Bn * Nn), 256, 0, stream>>>(l3, pos, buf1, Bn, Cf, 16, 48, 48, Nn, CT, Co);
    copy_cols_k<<<eb((long)Bn * Nn * 3), 256, 0, stream>>>(buf1, CT, Co + Cf, pos, 3, (long)Bn * Nn, 3);
    unpool_k<<<eb((long)Bn * 11000 * CT), 256, 0, stream>>>(buf1, upidx[0], upw[0], buf0, Bn, 11000, Nn, CT);
  }
  // ---- level 5: N=11000, 67 -> 32, edge0, then GCout -> 3 ----
  {
    int Nn = 11000, C = 67, Co = 32, E = 6 * Nn;
    run_cheb(stream, buf0, Bn, Nn, C, edge[0], E, gcw[4], gcb[4], Co, buf1, buf2, deg, nrm, wgt);
    instnorm_relu_k<<<eb((long)Bn * Nn), 256, 0, stream>>>(buf2, (long)Bn * Nn, Co);
    // GCout: ChebConv K=6, 32 -> 3, no bias, no norm -> d_out[0:132000]
    run_cheb(stream, buf2, Bn, Nn, Co, edge[0], E, gcout_w, nullptr, 3, buf1, out, deg, nrm, wgt);
  }
}
